// ProteinGraphNetwork_3582002725363
// MI455X (gfx1250) — compile-verified, matrix operand order checked by reading, not run on hardware
//
#include <hip/hip_runtime.h>

#define BS 2
#define N  384
#define H  64
#define NT (N/16)
#define TILES (BS*N*NT)

typedef __attribute__((ext_vector_type(16))) _Float16 v16h;
typedef __attribute__((ext_vector_type(8)))  _Float16 v8h;
typedef __attribute__((ext_vector_type(8)))  float    v8f;

// ---------------------------------------------------------------- utilities
__global__ void zero_kernel(float* __restrict__ p, int n) {
  int i = blockIdx.x * blockDim.x + threadIdx.x;
  if (i < n) p[i] = 0.f;
}

// Y[r][n] = bias[n] + X[r][:K] . W[:K][n]   (W row-major [K][64])
__global__ void rowgemm_kernel(const float* __restrict__ X, int ldx, int K,
                               const float* __restrict__ W, const float* __restrict__ bias,
                               float* __restrict__ Y, int rows) {
  int idx = blockIdx.x * blockDim.x + threadIdx.x;
  if (idx >= rows * H) return;
  int r = idx >> 6, n = idx & 63;
  float s = bias ? bias[n] : 0.f;
  const float* xr = X + (size_t)r * ldx;
  for (int k = 0; k < K; ++k) s += xr[k] * W[(size_t)k * H + n];
  Y[idx] = s;
}

// Pack 64x64 f32 weight block into WMMA B-fragment lane layout (f16).
// frag = t*2 + kc : t = N-tile (0..3), kc = K-chunk (0..1).
// Lane l (l<16): col n = t*16+l, K = kc*32 + {0..7, 16..23}
// Lane l (l>=16): col n = t*16+(l-16), K = kc*32 + {8..15, 24..31}
__global__ void pack_w_kernel(const float* __restrict__ W, _Float16* __restrict__ Wp) {
  int tid = threadIdx.x;            // 256 threads == 8 frags * 32 lanes
  int frag = tid >> 5, lane = tid & 31;
  int t = frag >> 1, kc = frag & 1;
  int n = t * 16 + (lane & 15);
  int kb2 = (lane & 16) ? 8 : 0;
  for (int h = 0; h < 16; ++h) {
    int k = kc * 32 + kb2 + ((h & 8) ? 16 : 0) + (h & 7);
    Wp[(frag * 32 + lane) * 16 + h] = (_Float16)W[k * H + n];
  }
}

// ---------------------------------------------------------------- WMMA edge tile
struct TileCtx {
  v8f acc[4];
  int b, i, j0, ncol, mb;
  size_t base;   // edge-row base index
};

__device__ inline void edge_tile_compute(const _Float16* __restrict__ Ain,
                                         const _Float16* __restrict__ Wp,
                                         const float* __restrict__ P,
                                         const float* __restrict__ Q,
                                         const float* __restrict__ RB,
                                         int wave, int lane, TileCtx& c) {
  int b = wave / (N * NT);
  int rem = wave % (N * NT);
  int i = rem / NT;
  int j0 = (rem % NT) * 16;
  c.b = b; c.i = i; c.j0 = j0;
  c.ncol = lane & 15;
  c.mb = (lane >> 4) * 8;
  size_t base = ((size_t)(b * N + i)) * N + j0;
  c.base = base;

  // A fragments: row = j0 + (lane&15); halves 0..7 -> K kb2+0..7, halves 8..15 -> K kb2+16..23
  int mrow = lane & 15;
  int kb2 = (lane & 16) ? 8 : 0;
  const _Float16* ar = Ain + (base + mrow) * H;
  union { v16h v; v8h h[2]; } ua0, ua1;
  ua0.h[0] = *(const v8h*)(ar + kb2);
  ua0.h[1] = *(const v8h*)(ar + kb2 + 16);
  ua1.h[0] = *(const v8h*)(ar + 32 + kb2);
  ua1.h[1] = *(const v8h*)(ar + 32 + kb2 + 16);

  // B fragments: pre-packed, contiguous per lane
  v16h bf[8];
#pragma unroll
  for (int f = 0; f < 8; ++f) bf[f] = *(const v16h*)(Wp + (f * 32 + lane) * 16);

  const float* Prow  = P + ((size_t)(b * N + i)) * H;
  const float* RBrow = RB + b * H;
#pragma unroll
  for (int t = 0; t < 4; ++t) {
    int n = t * 16 + c.ncol;
    float pn = Prow[n] + RBrow[n];
    v8f a;
#pragma unroll
    for (int p = 0; p < 8; ++p)
      a[p] = pn + Q[((size_t)(b * N) + j0 + c.mb + p) * H + n];
    a = __builtin_amdgcn_wmma_f32_16x16x32_f16(false, ua0.v, false, bf[t * 2 + 0],
                                               (short)0, a, false, false);
    a = __builtin_amdgcn_wmma_f32_16x16x32_f16(false, ua1.v, false, bf[t * 2 + 1],
                                               (short)0, a, false, false);
    c.acc[t] = a;
  }
}

// A2 = relu(A @ WeA + P_i + Q_j + rb); Aout = (residual ? A + A2 : A2);
// esum[b,i,:] += sum_j A2   (for node aggregation / global mean)
__global__ void __launch_bounds__(256)
edge_wmma_kernel(const _Float16* __restrict__ Ain, const _Float16* __restrict__ Wp,
                 const float* __restrict__ P, const float* __restrict__ Q,
                 const float* __restrict__ RB, _Float16* __restrict__ Aout,
                 float* __restrict__ esum, int residual) {
  int gtid = blockIdx.x * blockDim.x + threadIdx.x;
  int wave = gtid >> 5, lane = threadIdx.x & 31;
  if (wave >= TILES) return;
  TileCtx c;
  edge_tile_compute(Ain, Wp, P, Q, RB, wave, lane, c);
#pragma unroll
  for (int t = 0; t < 4; ++t) {
    int n = t * 16 + c.ncol;
    float s = 0.f;
#pragma unroll
    for (int p = 0; p < 8; ++p) {
      size_t idx = (c.base + c.mb + p) * H + n;
      float r = fmaxf(c.acc[t][p], 0.f);
      s += r;
      float o = r;
      if (residual) o += (float)Ain[idx];
      Aout[idx] = (_Float16)o;
    }
    atomicAdd(&esum[((size_t)(c.b * N + c.i)) * H + n], s);
  }
}

// Decode edge block fused with final linear head: out[b,i,j] = relu(...)[:] . out_W + out_b
__global__ void __launch_bounds__(256)
decode_edge_kernel(const _Float16* __restrict__ Ain, const _Float16* __restrict__ Wp,
                   const float* __restrict__ P, const float* __restrict__ Q,
                   const float* __restrict__ RB, const float* __restrict__ outW,
                   const float* __restrict__ outb, float* __restrict__ out) {
  int gtid = blockIdx.x * blockDim.x + threadIdx.x;
  int wave = gtid >> 5, lane = threadIdx.x & 31;
  if (wave >= TILES) return;
  TileCtx c;
  edge_tile_compute(Ain, Wp, P, Q, RB, wave, lane, c);
  float rp[8] = {0.f, 0.f, 0.f, 0.f, 0.f, 0.f, 0.f, 0.f};
#pragma unroll
  for (int t = 0; t < 4; ++t) {
    float w = outW[t * 16 + c.ncol];
#pragma unroll
    for (int p = 0; p < 8; ++p) rp[p] += fmaxf(c.acc[t][p], 0.f) * w;
  }
#pragma unroll
  for (int p = 0; p < 8; ++p) {
    rp[p] += __shfl_xor(rp[p], 1, 32);
    rp[p] += __shfl_xor(rp[p], 2, 32);
    rp[p] += __shfl_xor(rp[p], 4, 32);
    rp[p] += __shfl_xor(rp[p], 8, 32);
  }
  if (c.ncol == 0) {      // lanes 0 (rows 0..7) and 16 (rows 8..15)
    float bb = outb[0];
    float* orow = out + ((size_t)(c.b * N + c.i)) * N + c.j0 + c.mb;
#pragma unroll
    for (int p = 0; p < 8; ++p) orow[p] = rp[p] + bb;
  }
}

// ---------------------------------------------------------------- encode edge (K=1 scalar distance)
__global__ void encode_edge_kernel(const float* __restrict__ x, const float* __restrict__ P,
                                   const float* __restrict__ Q, const float* __restrict__ RB,
                                   const float* __restrict__ We /*66x64, row0 = dist weights*/,
                                   _Float16* __restrict__ Aout, float* __restrict__ esum) {
  int blk = blockIdx.x;                     // (b, i, j-chunk of 64)
  int b = blk / (N * (N / 64));
  int rem = blk % (N * (N / 64));
  int i = rem / (N / 64);
  int j0 = (rem % (N / 64)) * 64;
  __shared__ float xi[32], d[64], pn[64], w0[64];
  int t = threadIdx.x;
  if (t < 32) xi[t] = x[((size_t)(b * N + i)) * 32 + t];
  if (t >= 64 && t < 128)  { int n = t - 64;  pn[n] = P[(b * N + i) * H + n] + RB[b * H + n]; }
  if (t >= 128 && t < 192) { int n = t - 128; w0[n] = We[n]; }
  __syncthreads();
  if (t < 64) {
    int j = j0 + t;
    const float* xj = x + ((size_t)(b * N + j)) * 32;
    float s = 0.f;
    for (int k = 0; k < 32; ++k) { float df = xi[k] - xj[k]; s += df * df; }
    d[t] = sqrtf(fmaxf(s, 1e-12f));
  }
  __syncthreads();
  int n = t & 63, jb = t >> 6;              // 4 j-groups x 64 outputs
  float accs = 0.f;
  for (int cc = 0; cc < 16; ++cc) {
    int jl = jb * 16 + cc;
    float val = fmaxf(d[jl] * w0[n] + pn[n] + Q[((size_t)(b * N) + j0 + jl) * H + n], 0.f);
    accs += val;
    Aout[(((size_t)(b * N + i)) * N + j0 + jl) * H + n] = (_Float16)val;
  }
  atomicAdd(&esum[(b * N + i) * H + n], accs);
}

// ---------------------------------------------------------------- node & global updates
__global__ void node_kernel(const float* __restrict__ Vin, int vdim,
                            const float* __restrict__ esum, const float* __restrict__ u, int gdim,
                            const float* __restrict__ Wv, const float* __restrict__ bv,
                            float* __restrict__ V2, float* __restrict__ Vout,
                            int residual, float invn) {
  int idx = blockIdx.x * blockDim.x + threadIdx.x;
  if (idx >= BS * N * H) return;
  int r = idx >> 6, n = idx & 63, b = r / N;
  float s = bv[n];
  const float* vr = Vin + (size_t)r * vdim;
  for (int k = 0; k < vdim; ++k) s += vr[k] * Wv[(size_t)k * H + n];
  const float* er = esum + (size_t)r * H;
  for (int k = 0; k < H; ++k) s += er[k] * invn * Wv[(size_t)(vdim + k) * H + n];
  for (int g = 0; g < gdim; ++g) s += u[b * H + g] * Wv[(size_t)(vdim + H + g) * H + n];
  float v2 = fmaxf(s, 0.f);
  V2[idx] = v2;
  Vout[idx] = residual ? vr[n] + v2 : v2;
}

__global__ void global_kernel(const float* __restrict__ u, int gdim,
                              const float* __restrict__ V2, const float* __restrict__ esum,
                              const float* __restrict__ Wu, const float* __restrict__ bu,
                              float* __restrict__ uout, int residual) {
  int b = blockIdx.x, t = threadIdx.x;      // 64 threads per batch
  __shared__ float vm[64], am[64], uc[64];
  float sv = 0.f, sa = 0.f;
  for (int i = 0; i < N; ++i) {
    sv += V2[((size_t)(b * N + i)) * H + t];
    sa += esum[((size_t)(b * N + i)) * H + t];
  }
  vm[t] = sv / (float)N;
  am[t] = sa / ((float)N * (float)N);
  if (t < gdim) uc[t] = u[b * H + t];
  __syncthreads();
  float s = bu[t];
  for (int g = 0; g < gdim; ++g) s += uc[g] * Wu[(size_t)g * H + t];
  for (int k = 0; k < H; ++k)  s += vm[k] * Wu[(size_t)(gdim + k) * H + t];
  for (int k = 0; k < H; ++k)  s += am[k] * Wu[(size_t)(gdim + H + k) * H + t];
  float r = fmaxf(s, 0.f);
  uout[b * H + t] = residual ? u[b * H + t] + r : r;
}

// ---------------------------------------------------------------- host orchestration
extern "C" void kernel_launch(void* const* d_in, const int* in_sizes, int n_in,
                              void* d_out, int out_size, void* d_ws, size_t ws_size,
                              hipStream_t stream) {
  const float* x       = (const float*)d_in[0];
  const float* enc_We  = (const float*)d_in[2];
  const float* enc_be  = (const float*)d_in[3];
  const float* enc_Wv  = (const float*)d_in[4];
  const float* enc_bv  = (const float*)d_in[5];
  const float* enc_Wu  = (const float*)d_in[6];
  const float* enc_bu  = (const float*)d_in[7];
  const float* proc_We = (const float*)d_in[8];
  const float* proc_be = (const float*)d_in[9];
  const float* proc_Wv = (const float*)d_in[10];
  const float* proc_bv = (const float*)d_in[11];
  const float* proc_Wu = (const float*)d_in[12];
  const float* proc_bu = (const float*)d_in[13];
  const float* dec_We  = (const float*)d_in[14];
  const float* dec_be  = (const float*)d_in[15];
  const float* out_W   = (const float*)d_in[20];
  const float* out_b   = (const float*)d_in[21];
  float* out = (float*)d_out;

  char* ws = (char*)d_ws;
  size_t off = 0;
  auto alloc = [&](size_t bytes) -> char* {
    char* p = ws + off;
    off += (bytes + 255) & ~(size_t)255;
    return p;
  };
  _Float16* A0  = (_Float16*)alloc((size_t)BS * N * N * H * 2);
  _Float16* A1  = (_Float16*)alloc((size_t)BS * N * N * H * 2);
  float* P      = (float*)alloc((size_t)BS * N * H * 4);
  float* Q      = (float*)alloc((size_t)BS * N * H * 4);
  float* RB     = (float*)alloc((size_t)BS * H * 4);
  float* esum   = (float*)alloc((size_t)BS * N * H * 4);
  float* V2t    = (float*)alloc((size_t)BS * N * H * 4);
  float* Vc     = (float*)alloc((size_t)BS * N * H * 4);
  float* Vn     = (float*)alloc((size_t)BS * N * H * 4);
  float* U0     = (float*)alloc((size_t)BS * H * 4);
  float* U1     = (float*)alloc((size_t)BS * H * 4);
  _Float16* Wpk = (_Float16*)alloc(8 * 32 * 16 * 2);

  const float invn = 1.0f / (float)N;
  const int nvH = BS * N * H;                     // 49152
  const int gB  = (nvH + 255) / 256;
  const int edgeBlocks = TILES / 8;               // 8 waves per 256-thread block

  // u0 = 0 (initial global state, gdim=1)
  zero_kernel<<<1, 256, 0, stream>>>(U0, BS * H);

  // ---- ENCODE ----
  rowgemm_kernel<<<gB, 256, 0, stream>>>(x, 32, 32, enc_We + 1 * H,  nullptr, P, BS * N);
  rowgemm_kernel<<<gB, 256, 0, stream>>>(x, 32, 32, enc_We + 33 * H, nullptr, Q, BS * N);
  rowgemm_kernel<<<1, 256, 0, stream>>>(U0, H, 1, enc_We + 65 * H, enc_be, RB, BS);
  zero_kernel<<<gB, 256, 0, stream>>>(esum, nvH);
  encode_edge_kernel<<<BS * N * (N / 64), 256, 0, stream>>>(x, P, Q, RB, enc_We, A0, esum);
  node_kernel<<<gB, 256, 0, stream>>>(x, 32, esum, U0, 1, enc_Wv, enc_bv, V2t, Vc, 0, invn);
  global_kernel<<<BS, 64, 0, stream>>>(U0, 1, V2t, esum, enc_Wu, enc_bu, U1, 0);

  // ---- PROCESS (3 residual blocks, ping-pong) ----
  _Float16 *Ain = A0, *Aout = A1;
  float *Vin = Vc, *Vout = Vn;
  float *Uin = U1, *Uout = U0;
  for (int it = 0; it < 3; ++it) {
    const float* We = proc_We + (size_t)it * 256 * H;
    const float* be = proc_be + it * H;
    const float* Wv = proc_Wv + (size_t)it * 192 * H;
    const float* bv = proc_bv + it * H;
    const float* Wu = proc_Wu + (size_t)it * 192 * H;
    const float* bu = proc_bu + it * H;
    pack_w_kernel<<<1, 256, 0, stream>>>(We, Wpk);
    rowgemm_kernel<<<gB, 256, 0, stream>>>(Vin, H, H, We + 64 * H,  nullptr, P, BS * N);
    rowgemm_kernel<<<gB, 256, 0, stream>>>(Vin, H, H, We + 128 * H, nullptr, Q, BS * N);
    rowgemm_kernel<<<1, 256, 0, stream>>>(Uin, H, H, We + 192 * H, be, RB, BS);
    zero_kernel<<<gB, 256, 0, stream>>>(esum, nvH);
    edge_wmma_kernel<<<edgeBlocks, 256, 0, stream>>>(Ain, Wpk, P, Q, RB, Aout, esum, 1);
    node_kernel<<<gB, 256, 0, stream>>>(Vin, H, esum, Uin, H, Wv, bv, V2t, Vout, 1, invn);
    global_kernel<<<BS, 64, 0, stream>>>(Uin, H, V2t, esum, Wu, bu, Uout, 1);
    _Float16* ta = Ain; Ain = Aout; Aout = ta;
    float* tv = Vin; Vin = Vout; Vout = tv;
    float* tu = Uin; Uin = Uout; Uout = tu;
  }

  // ---- DECODE (only edge output feeds the head; node/global updates are dead) ----
  pack_w_kernel<<<1, 256, 0, stream>>>(dec_We, Wpk);
  rowgemm_kernel<<<gB, 256, 0, stream>>>(Vin, H, H, dec_We + 64 * H,  nullptr, P, BS * N);
  rowgemm_kernel<<<gB, 256, 0, stream>>>(Vin, H, H, dec_We + 128 * H, nullptr, Q, BS * N);
  rowgemm_kernel<<<1, 256, 0, stream>>>(Uin, H, H, dec_We + 192 * H, dec_be, RB, BS);
  decode_edge_kernel<<<edgeBlocks, 256, 0, stream>>>(Ain, Wpk, P, Q, RB, out_W, out_b, out);
}
